// TrajectoryEncoder_9216999817928
// MI455X (gfx1250) — matrix operand order repaired and checked by reading, not compiled, not run on hardware
//
#include <hip/hip_runtime.h>
#include <hip/hip_bf16.h>

// Problem dimensions (fixed by the reference).
#define T_STEPS 8192
#define D_IN    512
#define H_DIM   1024
#define E_OUT   512
#define FOURH   4096
#define NWG     64      // persistent workgroups for the scan

typedef __attribute__((ext_vector_type(16))) __bf16 v16bf;
typedef __attribute__((ext_vector_type(8)))  __bf16 v8bf;
typedef __attribute__((ext_vector_type(4)))  __bf16 v4bf;
typedef __attribute__((ext_vector_type(8)))  float  v8f;

// ---------------------------------------------------------------------------
// Prep: fp32 -> bf16 elementwise convert (trajectory, Wh)
// ---------------------------------------------------------------------------
__global__ void k_f32_to_bf16(const float* __restrict__ src,
                              __bf16* __restrict__ dst, int n) {
  int i = blockIdx.x * blockDim.x + threadIdx.x;
  if (i < n) dst[i] = (__bf16)src[i];
}

// ---------------------------------------------------------------------------
// Prep: pack Wi [512 x 4096] fp32 row-major into the WMMA 32x16 bf16 B-tile
// per-lane layout:  Bp[n_tile(256)][k_chunk(16)][lane(32)][elem(16)]
//   lane<16 : holds B[k=kc*32+0..7,16..23,  n=nt*16+lane]
//   lane>=16: holds B[k=kc*32+8..15,24..31, n=nt*16+lane-16]
// so the GEMM reads one contiguous, coalesced 32B v16bf per lane per chunk.
// ---------------------------------------------------------------------------
__global__ void k_pack_wi(const float* __restrict__ Wi, __bf16* __restrict__ Bp) {
  int gid  = blockIdx.x * blockDim.x + threadIdx.x;   // 256*16*32 threads
  int lane = gid & 31;
  int kc   = (gid >> 5) & 15;
  int nt   = gid >> 9;
  int n    = nt * 16 + (lane & 15);
  int kb   = kc * 32 + ((lane & 16) ? 8 : 0);
  __bf16* out = Bp + (size_t)gid * 16;
#pragma unroll
  for (int e = 0; e < 16; ++e)
    out[e] = (__bf16)Wi[(size_t)(kb + (e < 8 ? e : e + 8)) * FOURH + n];
}

// ---------------------------------------------------------------------------
// Prep: zero h (double-buffered), c, barrier state — every launch (graph-safe)
// ---------------------------------------------------------------------------
__global__ void k_init(float* h_g, float* c_g, unsigned* bar) {
  int i = blockIdx.x * blockDim.x + threadIdx.x;
  if (i < 2 * H_DIM) h_g[i] = 0.f;
  if (i < H_DIM)     c_g[i] = 0.f;
  if (i < 2)         bar[i] = 0u;
}

// ---------------------------------------------------------------------------
// Phase 1: xz = trajectory @ Wi + b  via v_wmma_f32_16x16x32_bf16.
// One 16x16 output tile per wave; K=512 in 16 chunks of 32.
// ---------------------------------------------------------------------------
__global__ __launch_bounds__(256) void k_gemm_xz(
    const __bf16* __restrict__ A,    // [8192,512] bf16 row-major
    const __bf16* __restrict__ Bp,   // packed Wi (see k_pack_wi)
    const float*  __restrict__ bias, // [4096]
    float* __restrict__ C)           // [8192,4096] fp32
{
  int wave = (blockIdx.x * blockDim.x + threadIdx.x) >> 5;
  int lane = threadIdx.x & 31;
  int mt = wave >> 8;         // 512 m-tiles
  int nt = wave & 255;        // 256 n-tiles

  // A layout (16-bit 16x32): lane 0-15 row M=lane, K=0..7 & 16..23;
  //                          lane 16-31 row M=lane-16, K=8..15 & 24..31.
  int mrow = mt * 16 + (lane & 15);
  int kb   = (lane & 16) ? 8 : 0;
  const __bf16* arow = A + (size_t)mrow * D_IN;
  const __bf16* bp   = Bp + ((size_t)nt * 16 * 32 + lane) * 16;

  v8f acc = {};
#pragma unroll
  for (int kc = 0; kc < 16; ++kc) {
    int k0 = kc * 32;
    v8bf alo = *(const v8bf*)(arow + k0 + kb);        // K = k0+kb .. +7
    v8bf ahi = *(const v8bf*)(arow + k0 + kb + 16);   // K = k0+kb+16 .. +23
    v16bf a = __builtin_shufflevector(alo, ahi,
        0, 1, 2, 3, 4, 5, 6, 7, 8, 9, 10, 11, 12, 13, 14, 15);
    v16bf b = *(const v16bf*)(bp + (size_t)kc * 32 * 16);
    acc = __builtin_amdgcn_wmma_f32_16x16x32_bf16(
        false, a, false, b, (short)0, acc, false, false);
  }

  // C layout: lanes 0-15 -> M = e, lanes 16-31 -> M = e+8; N = lane&15.
  int ncol  = nt * 16 + (lane & 15);
  float bv  = bias[ncol];
  int rbase = mt * 16 + ((lane & 16) ? 8 : 0);
#pragma unroll
  for (int e = 0; e < 8; ++e)
    C[(size_t)(rbase + e) * FOURH + ncol] = acc[e] + bv;
}

// ---------------------------------------------------------------------------
// Device-wide split barrier (generation counter) through L2 atomics.
// ---------------------------------------------------------------------------
__device__ __forceinline__ void grid_barrier(unsigned* bar, unsigned nwg) {
  __syncthreads();
  if (threadIdx.x == 0) {
    __threadfence();  // publish h writes device-wide before arriving
    unsigned gen  = __hip_atomic_load(&bar[1], __ATOMIC_RELAXED,
                                      __HIP_MEMORY_SCOPE_AGENT);
    unsigned prev = __hip_atomic_fetch_add(&bar[0], 1u, __ATOMIC_ACQ_REL,
                                           __HIP_MEMORY_SCOPE_AGENT);
    if (prev == nwg - 1) {
      __hip_atomic_store(&bar[0], 0u, __ATOMIC_RELAXED,
                         __HIP_MEMORY_SCOPE_AGENT);
      __hip_atomic_store(&bar[1], gen + 1, __ATOMIC_RELEASE,
                         __HIP_MEMORY_SCOPE_AGENT);
    } else {
      while (__hip_atomic_load(&bar[1], __ATOMIC_ACQUIRE,
                               __HIP_MEMORY_SCOPE_AGENT) == gen)
        __builtin_amdgcn_s_sleep(1);
    }
  }
  __syncthreads();
}

__device__ __forceinline__ float sigmoidf_(float x) {
  return 1.f / (1.f + __expf(-x));
}

// ---------------------------------------------------------------------------
// Phase 2: persistent LSTM scan. WG w owns hidden units [16w, 16w+16):
//   64 gate columns j = q*1024 + 16w + idx, q in {i,f,g,o}.
// Vectorized inner loop: each lane loads b64 = 4 consecutive bf16 columns
// and accumulates 4 outputs; 16 lanes cover all 64 outputs, the two
// half-waves take different k-sub-segments -> 16 k-segments of 64.
// Per step per WG: 512 load instructions x 256B/wave (vs 2048 x 64B before).
// h is double-buffered in global so step t reads never race t+1 writes.
// Phase 3 (tail): out = c @ W_out + b_out, 8 outputs per WG, wave-reduced.
// ---------------------------------------------------------------------------
__global__ __launch_bounds__(256) void k_lstm_scan(
    const float*  __restrict__ xz,    // [8192,4096]
    const __bf16* __restrict__ Whb,   // [1024,4096] bf16
    const float*  __restrict__ Wout,  // [1024,512]
    const float*  __restrict__ bout,  // [512]
    float* h_g,                       // [2][1024]
    float* c_g,                       // [1024]
    float* out,                       // [512]
    unsigned* bar)
{
  __shared__ float h_s[H_DIM];
  __shared__ float partial[16][64];
  __shared__ float zbuf[64];
  __shared__ float c_s[16];

  const int tid  = threadIdx.x;
  const int w    = blockIdx.x;
  const int wave = tid >> 5, lane = tid & 31;
  const int half = lane >> 4;                 // 0/1: k sub-segment
  const int L    = lane & 15;                 // output group (4 outputs each)
  const int seg  = wave * 2 + half;           // 0..15, 64 k-values each
  const int q    = L >> 2;                    // gate index 0..3
  const int j0   = q * H_DIM + w * 16 + (L & 3) * 4;  // 4 consecutive columns
  const __bf16* colp = Whb + (size_t)(seg * 64) * FOURH + j0;

  if (tid < 16) c_s[tid] = 0.f;

  for (int t = 0; t < T_STEPS; ++t) {
    // Broadcast current h into LDS; prefetch this step's xz row.
    const float* hsrc = h_g + (t & 1) * H_DIM;
    __builtin_prefetch(&xz[(size_t)t * FOURH + tid * 16], 0, 0);
    for (int i = tid; i < H_DIM; i += 256) h_s[i] = hsrc[i];
    __syncthreads();

    // 4-output partial dot over this half-wave's 64-k segment.
    float a0 = 0.f, a1 = 0.f, a2 = 0.f, a3 = 0.f;
    const float* hk = h_s + seg * 64;
#pragma unroll 8
    for (int k = 0; k < 64; ++k) {
      v4bf wv = *(const v4bf*)(colp + (size_t)k * FOURH);  // 8B, coalesced
      float hv = hk[k];
      a0 = fmaf((float)wv[0], hv, a0);
      a1 = fmaf((float)wv[1], hv, a1);
      a2 = fmaf((float)wv[2], hv, a2);
      a3 = fmaf((float)wv[3], hv, a3);
    }
    partial[seg][L * 4 + 0] = a0;
    partial[seg][L * 4 + 1] = a1;
    partial[seg][L * 4 + 2] = a2;
    partial[seg][L * 4 + 3] = a3;
    __syncthreads();

    if (tid < 64) {
      float z = xz[(size_t)t * FOURH + (tid >> 4) * H_DIM + w * 16 + (tid & 15)];
#pragma unroll
      for (int s = 0; s < 16; ++s) z += partial[s][tid];
      zbuf[tid] = z;
    }
    __syncthreads();

    if (tid < 16) {
      float ig = sigmoidf_(zbuf[tid]);
      float fg = sigmoidf_(zbuf[16 + tid]);
      float gg = tanhf(zbuf[32 + tid]);
      float og = sigmoidf_(zbuf[48 + tid]);
      float cn = fg * c_s[tid] + ig * gg;
      c_s[tid] = cn;
      h_g[((t + 1) & 1) * H_DIM + w * 16 + tid] = og * tanhf(cn);
    }
    grid_barrier(bar, NWG);
  }

  // Publish final cell state, then project: out = c @ W_out + b_out.
  if (tid < 16) c_g[w * 16 + tid] = c_s[tid];
  grid_barrier(bar, NWG);

  int e = w * 8 + wave;                 // one output column per wave
  float a = 0.f;
  for (int k = lane; k < H_DIM; k += 32)
    a = fmaf(c_g[k], Wout[(size_t)k * E_OUT + e], a);
#pragma unroll
  for (int off = 16; off > 0; off >>= 1)
    a += __shfl_down(a, off, 32);
  if (lane == 0) out[e] = a + bout[e];
}

// ---------------------------------------------------------------------------
extern "C" void kernel_launch(void* const* d_in, const int* in_sizes, int n_in,
                              void* d_out, int out_size, void* d_ws,
                              size_t ws_size, hipStream_t stream) {
  (void)in_sizes; (void)n_in; (void)out_size; (void)ws_size;
  const float* traj = (const float*)d_in[0];   // [8192,512]
  const float* Wi   = (const float*)d_in[1];   // [512,4096]
  const float* Wh   = (const float*)d_in[2];   // [1024,4096]
  const float* b    = (const float*)d_in[3];   // [4096]
  const float* Wout = (const float*)d_in[4];   // [1024,512]
  const float* bout = (const float*)d_in[5];   // [512]
  float* out = (float*)d_out;                  // [512]

  char* ws = (char*)d_ws;
  size_t off = 0;
  auto alloc = [&](size_t bytes) -> char* {
    char* p = ws + off;
    off = (off + bytes + 255) & ~(size_t)255;
    return p;
  };
  __bf16*   A_b  = (__bf16*)alloc((size_t)T_STEPS * D_IN * 2);   //   8 MiB
  __bf16*   Bp   = (__bf16*)alloc((size_t)D_IN * FOURH * 2);     //   4 MiB
  __bf16*   Wh_b = (__bf16*)alloc((size_t)H_DIM * FOURH * 2);    //   8 MiB
  float*    xz   = (float*)alloc((size_t)T_STEPS * FOURH * 4);   // 128 MiB
  float*    h_g  = (float*)alloc(2 * H_DIM * 4);
  float*    c_g  = (float*)alloc(H_DIM * 4);
  unsigned* bar  = (unsigned*)alloc(256);

  k_f32_to_bf16<<<(T_STEPS * D_IN + 255) / 256, 256, 0, stream>>>(
      traj, A_b, T_STEPS * D_IN);
  k_pack_wi<<<(256 * 16 * 32) / 256, 256, 0, stream>>>(Wi, Bp);
  k_f32_to_bf16<<<(H_DIM * FOURH + 255) / 256, 256, 0, stream>>>(
      Wh, Wh_b, H_DIM * FOURH);
  k_init<<<(2 * H_DIM + 255) / 256, 256, 0, stream>>>(h_g, c_g, bar);

  // Phase 1: 512x256 16x16 tiles, one per wave, 8 waves per block.
  k_gemm_xz<<<16384, 256, 0, stream>>>(A_b, Bp, b, xz);

  // Phases 2+3: persistent scan with device-wide barrier.
  k_lstm_scan<<<NWG, 256, 0, stream>>>(xz, Wh_b, Wout, bout, h_g, c_g, out, bar);
}
